// Attention_7653631721648
// MI455X (gfx1250) — matrix operand order repaired and checked by reading, not compiled, not run on hardware
//
#include <hip/hip_runtime.h>
#include <hip/hip_bf16.h>

#define B_DIM 512
#define T_DIM 512
#define DV    256
#define DQ    128
#define DH    64

typedef __attribute__((ext_vector_type(16))) __bf16 v16bf;
typedef __attribute__((ext_vector_type(8)))  float  v8f;
typedef __attribute__((ext_vector_type(4)))  unsigned int u32x4;
typedef __attribute__((ext_vector_type(8)))  int          i32x8;
typedef __attribute__((ext_vector_type(4)))  int          i32x4;

__device__ __forceinline__ unsigned short f2bfbits(float f) {
    union { __bf16 b; unsigned short s; } cv;
    cv.b = (__bf16)f;                  // native v_cvt on gfx1250
    return cv.s;
}

__device__ __forceinline__ float fast_tanh(float x) {
    // tanh(x) = 1 - 2/(exp(2x)+1); saturates correctly at +-1, ~5 VALU via v_exp_f32
    float e = __expf(2.0f * x);
    return 1.0f - 2.0f / (e + 1.0f);
}

union FragBF {
    v16bf          v;
    __bf16         h[16];
    uint4          q[2];
};

// ---------------------------------------------------------------------------
// Phase 1: qb[b][n] = query[b,:] . W2[n,:] + b2[n] + b1[n]      (tiny GEMM)
// ---------------------------------------------------------------------------
__global__ void qproj_kernel(const float* __restrict__ query,
                             const float* __restrict__ W2,
                             const float* __restrict__ b1,
                             const float* __restrict__ b2,
                             float* __restrict__ qb)
{
    const int b = blockIdx.x;
    const int n = threadIdx.x;            // 0..63
    float acc = b1[n] + b2[n];
    const float* qr = query + b * DQ;
    const float* wr = W2 + n * DQ;
#pragma unroll 8
    for (int q = 0; q < DQ; ++q) acc += qr[q] * wr[q];
    qb[b * DH + n] = acc;
}

// ---------------------------------------------------------------------------
// Phase 2: big GEMM [B*T,256]x[256,64] via v_wmma_f32_16x16x32_bf16.
// The 128x256 f32 values tile is DMA'd into LDS by the Tensor Data Mover
// (tensor_load_to_lds, TENSORcnt), overlapped with W1 fragment staging.
// Fused epilogue: bias add, fast tanh, Wv-dot, score[b*T+t].
// Block: 256 threads (8 waves), 128 rows. LDS: 128KB f32 tile + 32KB W1 frags.
// ---------------------------------------------------------------------------
__global__ void __launch_bounds__(256)
attn_scores_wmma(const float* __restrict__ values,
                 const float* __restrict__ W1,
                 const float* __restrict__ Wv,
                 const float* __restrict__ bv,
                 const float* __restrict__ qb,
                 float* __restrict__ score)
{
    extern __shared__ char smem[];
    float*          sValF = (float*)smem;                      // [128][256] f32 = 131072 B
    unsigned short* sW    = (unsigned short*)(smem + 131072);  // [4][8][32][16] bf16 = 32768 B

    const int tid     = threadIdx.x;
    const int rowBase = blockIdx.x * 128;

    // ---- TDM: one 2D tensor DMA of the 128x256 f32 tile into LDS ----------
    // Scalar branch so only wave 0 issues the DMA (TDM ignores EXEC; the load
    // is idempotent anyway). D# per CDNA5 ISA 8.3/8.4.
    if (__builtin_amdgcn_readfirstlane((int)(tid >> 5)) == 0) {
        unsigned long long ga  = (unsigned long long)(const void*)(values + (size_t)rowBase * DV);
        unsigned int       lds = (unsigned int)(unsigned long long)(uintptr_t)sValF;
        u32x4 g0;
        g0[0] = 1u;                                    // count=1, no gather, user mode
        g0[1] = lds;                                   // lds_addr (bytes)
        g0[2] = (unsigned int)ga;                      // global_addr[31:0]
        g0[3] = (unsigned int)((ga >> 32) & 0x01FFFFFFull) | (2u << 30); // ga[56:32], type=2
        i32x8 g1;
        g1[0] = (int)(2u << 16);                       // wg_mask=0, data_size=4B
        g1[1] = (int)(256u << 16);                     // tensor_dim0 = 256 (low bits<<16)
        g1[2] = 0;                                     // tensor_dim0_hi=0, tensor_dim1[15:0]=0
        g1[3] = (int)((256u << 16) | 4u);              // tensor_dim1[31:16]=4 (262144), tile_dim0=256
        g1[4] = 128;                                   // tile_dim1=128, tile_dim2=0
        g1[5] = 256;                                   // tensor_dim0_stride = 256
        g1[6] = 0;                                     // stride hi / dim1_stride lo
        g1[7] = 0;                                     // dim1_stride hi
        i32x4 gz = {0, 0, 0, 0};                       // groups 2/3 unused (2D)
        i32x8 z8 = {0, 0, 0, 0, 0, 0, 0, 0};           // extra octet (clang-23 6-arg form)
        __builtin_amdgcn_tensor_load_to_lds(g0, g1, gz, gz, z8, 0);
    }

    // ---- Stage W1 (f32 [DH][DV]) pre-swizzled into WMMA B-fragment layout,
    //      overlapping the TDM transfer.
    // B is KxN (32x16): lane -> N = lane&15, K = (e&7) + (e>=8 ? 16 : 0) + (lane>=16 ? 8 : 0).
#pragma unroll
    for (int i = 0; i < 64; ++i) {
        int idx  = tid + i * 256;          // 0..16383
        int e    = idx & 15;
        int lane = (idx >> 4) & 31;
        int ks   = (idx >> 9) & 7;
        int nt   = (idx >> 12) & 3;
        int n    = nt * 16 + (lane & 15);                 // DH index
        int k    = ks * 32 + ((lane & 16) ? 8 : 0) + (e & 7) + ((e & 8) ? 16 : 0);  // DV index
        sW[idx] = f2bfbits(W1[n * DV + k]);
    }

    __builtin_amdgcn_s_wait_tensorcnt(0);
    __syncthreads();

    const int w   = tid >> 5;
    const int l   = tid & 31;
    const int lh  = l & 15;
    const int hi8 = (l & 16) ? 8 : 0;

    v8f z = {0.f, 0.f, 0.f, 0.f, 0.f, 0.f, 0.f, 0.f};
    v8f acc[4];
    acc[0] = z; acc[1] = z; acc[2] = z; acc[3] = z;

    // A fragment: lane -> M = l&15, chunk0: K = kb..kb+7, chunk1: K = kb+16..kb+23
    // with kb = 32*ks + (l>=16 ? 8 : 0)   (per ISA 16-bit A 16x32 layout)
    const int row = w * 16 + lh;
#pragma unroll
    for (int ks = 0; ks < 8; ++ks) {
        const float4* ap4 = (const float4*)&sValF[row * DV + ks * 32 + hi8];
        float4 f0 = ap4[0];                // K = kb .. kb+3
        float4 f1 = ap4[1];                // K = kb+4 .. kb+7
        float4 f2 = ap4[4];                // K = kb+16 .. kb+19
        float4 f3 = ap4[5];                // K = kb+20 .. kb+23
        FragBF a;
        a.h[0]  = (__bf16)f0.x;  a.h[1]  = (__bf16)f0.y;
        a.h[2]  = (__bf16)f0.z;  a.h[3]  = (__bf16)f0.w;
        a.h[4]  = (__bf16)f1.x;  a.h[5]  = (__bf16)f1.y;
        a.h[6]  = (__bf16)f1.z;  a.h[7]  = (__bf16)f1.w;
        a.h[8]  = (__bf16)f2.x;  a.h[9]  = (__bf16)f2.y;
        a.h[10] = (__bf16)f2.z;  a.h[11] = (__bf16)f2.w;
        a.h[12] = (__bf16)f3.x;  a.h[13] = (__bf16)f3.y;
        a.h[14] = (__bf16)f3.z;  a.h[15] = (__bf16)f3.w;
#pragma unroll
        for (int nt = 0; nt < 4; ++nt) {
            FragBF bf;
            const uint4* bp = (const uint4*)&sW[((nt * 8 + ks) * 32 + l) * 16];
            bf.q[0] = bp[0];
            bf.q[1] = bp[1];
            acc[nt] = __builtin_amdgcn_wmma_f32_16x16x32_bf16(
                false, a.v, false, bf.v, (short)0, acc[nt], false, false);
        }
    }

    // Epilogue: h = gemm + qb[b][n]; s = sum_n Wv[n]*tanh(h) + bv
    // C layout: VGPR j -> M = j + hi8, N = nt*16 + (l&15)
    const int   bidx = rowBase >> 9;       // 512 rows (T) per batch, blocks aligned
    const float bvv  = bv[0];
    float rsum[8];
#pragma unroll
    for (int j = 0; j < 8; ++j) rsum[j] = 0.f;

#pragma unroll
    for (int nt = 0; nt < 4; ++nt) {
        int   n  = nt * 16 + lh;
        float wv = Wv[n];
        float qv = qb[bidx * DH + n];
#pragma unroll
        for (int j = 0; j < 8; ++j) {
            rsum[j] += wv * fast_tanh(acc[nt][j] + qv);
        }
    }
    // reduce over the 16 lanes of each half (xor<=8 stays within a half)
#pragma unroll
    for (int j = 0; j < 8; ++j) {
#pragma unroll
        for (int m = 1; m < 16; m <<= 1)
            rsum[j] += __shfl_xor(rsum[j], m, 32);
    }
    if (lh == 0) {
#pragma unroll
        for (int j = 0; j < 8; ++j) {
            score[rowBase + w * 16 + hi8 + j] = rsum[j] + bvv;
        }
    }
}

// ---------------------------------------------------------------------------
// Phase 3: softmax over b (axis 0) for each t. One wave per t.
// ---------------------------------------------------------------------------
__global__ void softmax_over_b(const float* __restrict__ score,
                               float* __restrict__ aw)
{
    const int t = blockIdx.x;
    const int l = threadIdx.x;             // 0..31
    float v[16];
    float m = -3.0e38f;
#pragma unroll
    for (int i = 0; i < 16; ++i) {
        v[i] = score[(l + i * 32) * T_DIM + t];
        m = fmaxf(m, v[i]);
    }
#pragma unroll
    for (int s = 16; s >= 1; s >>= 1) m = fmaxf(m, __shfl_xor(m, s, 32));
    float sum = 0.f;
#pragma unroll
    for (int i = 0; i < 16; ++i) { v[i] = __expf(v[i] - m); sum += v[i]; }
#pragma unroll
    for (int s = 16; s >= 1; s >>= 1) sum += __shfl_xor(sum, s, 32);
    float inv = 1.0f / sum;
#pragma unroll
    for (int i = 0; i < 16; ++i) {
        aw[(l + i * 32) * T_DIM + t] = v[i] * inv;
    }
}

// ---------------------------------------------------------------------------
// Phase 4: context[t][v] = sum_b aw[b][t] * values[b][t][v]   (bandwidth-bound)
// ---------------------------------------------------------------------------
__global__ void __launch_bounds__(256)
context_reduce(const float* __restrict__ values,
               const float* __restrict__ aw,
               float* __restrict__ ctx)
{
    const int t = blockIdx.x;
    const int v = threadIdx.x;             // 0..255
    float acc = 0.f;
    for (int b = 0; b < B_DIM; ++b) {
        float w = aw[b * T_DIM + t];
        acc += w * values[((size_t)(b * T_DIM + t)) * DV + v];
    }
    ctx[t * DV + v] = acc;
}

// ---------------------------------------------------------------------------
extern "C" void kernel_launch(void* const* d_in, const int* in_sizes, int n_in,
                              void* d_out, int out_size, void* d_ws, size_t ws_size,
                              hipStream_t stream)
{
    const float* query  = (const float*)d_in[0];
    const float* values = (const float*)d_in[1];
    const float* W1     = (const float*)d_in[2];
    const float* b1     = (const float*)d_in[3];
    const float* W2     = (const float*)d_in[4];
    const float* b2     = (const float*)d_in[5];
    const float* Wv     = (const float*)d_in[6];
    const float* bv     = (const float*)d_in[7];

    float* ctx = (float*)d_out;            // [T][DV]  = 131072 floats
    float* aw  = ctx + T_DIM * DV;         // [B][T]   = 262144 floats

    float* qb    = (float*)d_ws;           // [B][DH]  = 32768 floats
    float* score = qb + B_DIM * DH;        // [B][T]   = 262144 floats

    const int smem_bytes = 131072 + 32768; // f32 tile + swizzled W1 = 160 KB

    (void)hipFuncSetAttribute((const void*)attn_scores_wmma,
                              hipFuncAttributeMaxDynamicSharedMemorySize,
                              smem_bytes);

    qproj_kernel<<<B_DIM, DH, 0, stream>>>(query, W2, b1, b2, qb);

    attn_scores_wmma<<<(B_DIM * T_DIM) / 128, 256, smem_bytes, stream>>>(
        values, W1, Wv, bv, qb, score);

    softmax_over_b<<<T_DIM, 32, 0, stream>>>(score, aw);

    context_reduce<<<T_DIM, DV, 0, stream>>>(values, aw, ctx);
}
